// SpatialCrossAttention_14353780703930
// MI455X (gfx1250) — matrix operand order repaired and checked by reading, not compiled, not run on hardware
//
#include <hip/hip_runtime.h>
#include <hip/hip_bf16.h>
#include <stdint.h>

typedef __attribute__((ext_vector_type(16))) _Float16 hvec16;
typedef __attribute__((ext_vector_type(8)))  _Float16 hvec8;
typedef __attribute__((ext_vector_type(8)))  float    fvec8;

#define NQ_    2500
#define EMBED_ 256
#define NH_    8
#define NL_    4
#define NP_    8
#define NCAMS_ 6
#define LTOT_  19560
#define KSTEPS 8   // K=256 / 32

// ---------------------------------------------------------------------------
// CDNA5 async copy: 32 bytes global -> LDS per lane (2x ASYNC_TO_LDS_B128).
// IOFFSET is added to BOTH the LDS and global addresses (ISA 08 section 4.4),
// so the same operands with offset:16 move the second 16B chunk.
// ---------------------------------------------------------------------------
__device__ __forceinline__ void async_copy32(unsigned lds_addr, const float* gsrc) {
  asm volatile(
      "global_load_async_to_lds_b128 %0, %1, off\n\t"
      "global_load_async_to_lds_b128 %0, %1, off offset:16"
      :: "v"(lds_addr), "v"(gsrc) : "memory");
}
__device__ __forceinline__ void wait_async0() {
  asm volatile("s_wait_asynccnt 0x0" ::: "memory");
}

// ---------------------------------------------------------------------------
// qsum = query + query_pos  (lets the GEMM A-path be a pure async byte copy)
// ---------------------------------------------------------------------------
__global__ void sca_add_qpos(const float* __restrict__ a, const float* __restrict__ b,
                             float* __restrict__ c, int n) {
  int i = blockIdx.x * 256 + threadIdx.x;
  if (i < n) c[i] = a[i] + b[i];
}

// ---------------------------------------------------------------------------
// Pack weight matrix W (Nrows x 256, fp32; B[k,n] = W[n,k]) into f16 WMMA-B
// fragment order (contiguous K, matching the A path: lanes 0-15 hold K 0..15, lanes 16-31 K 16..31)
// ---------------------------------------------------------------------------
__global__ void sca_pack_w(const float* __restrict__ W, _Float16* __restrict__ dst,
                           int Nrows) {
  int idx = blockIdx.x * blockDim.x + threadIdx.x;
  int total = (Nrows >> 4) * KSTEPS * 32;
  if (idx >= total) return;
  int lane = idx & 31;
  int ks   = (idx >> 5) & 7;
  int nt   = idx >> 8;
  int n    = nt * 16 + (lane & 15);
  int k0   = ks * 32 + (lane >> 4) * 16;
  const float* src = W + (size_t)n * 256 + k0;
  _Float16* d = dst + (size_t)idx * 16;
#pragma unroll
  for (int j = 0; j < 16; ++j) d[j] = (_Float16)src[j];
}

// ---------------------------------------------------------------------------
// K=256 GEMM: C[M,N] = A @ B + bias (+res). f16 WMMA, f32 accumulate.
// Block = 256 threads = 8 waves -> 64x64 tile. A tiles stream into LDS (fp32)
// via double-buffered GLOBAL_LOAD_ASYNC_TO_LDS_B128; converted to f16 during
// fragment assembly. Output to f32 (Cf) or f16 (Ch).
// ---------------------------------------------------------------------------
__global__ __launch_bounds__(256) void sca_gemm_wmma(
    const float* __restrict__ A, const _Float16* __restrict__ Bp,
    const float* __restrict__ bias, const float* __restrict__ res,
    float* __restrict__ Cf, _Float16* __restrict__ Ch, int M, int N) {
  // two fp32 A-tile buffers: 64 rows x 32 K, row stride 36 floats (144B, 16B-aligned,
  // conflict-free: {36*m mod 64} distinct over 16 lanes)
  __shared__ alignas(16) float sA[2][64 * 36];

  const int tid  = threadIdx.x;
  const int lane = tid & 31;
  const int wid  = tid >> 5;
  const int wm   = wid & 3;          // 4 wave-rows of 16
  const int wn   = wid >> 2;         // 2 wave-cols of 32
  const int rowBlk = blockIdx.x * 64;
  const int kh     = lane >> 4;      // K-half select (16-bit A layout)
  const int fragRowF = (wm * 16 + (lane & 15)) * 36;
  const int ntBase   = blockIdx.y * 4 + wn * 2;

  // async staging: thread t copies row t/4, float-chunk (t%4)*8 (32B)
  const int stgRow = tid >> 2;
  const int stgC   = (tid & 3) * 8;
  int gRow = rowBlk + stgRow;
  if (gRow >= M) gRow = M - 1;       // clamp: pad rows never stored
  const float* gsrc = A + (size_t)gRow * 256 + stgC;
  const unsigned ldsOff[2] = {
      (unsigned)(uintptr_t)&sA[0][stgRow * 36 + stgC],
      (unsigned)(uintptr_t)&sA[1][stgRow * 36 + stgC]};

  fvec8 acc0; fvec8 acc1;
#pragma unroll
  for (int i = 0; i < 8; ++i) { acc0[i] = 0.0f; acc1[i] = 0.0f; }

  async_copy32(ldsOff[0], gsrc);               // prefetch k-step 0

  for (int ks = 0; ks < KSTEPS; ++ks) {
    const int cur = ks & 1;
    wait_async0();                             // own async writes to cur done
    __syncthreads();                           // everyone's writes visible;
                                               // also fences last iter's reads
    if (ks + 1 < KSTEPS)                       // stream next tile while computing
      async_copy32(ldsOff[cur ^ 1], gsrc + (ks + 1) * 32);

    // ---- B fragments: two coalesced b128 loads each, pre-packed layout ----
    const _Float16* b0p = Bp + ((size_t)((ntBase + 0) * KSTEPS + ks) * 32 + lane) * 16;
    const _Float16* b1p = Bp + ((size_t)((ntBase + 1) * KSTEPS + ks) * 32 + lane) * 16;
    hvec8 b0lo = *(const hvec8*)(b0p);
    hvec8 b0hi = *(const hvec8*)(b0p + 8);
    hvec8 b1lo = *(const hvec8*)(b1p);
    hvec8 b1hi = *(const hvec8*)(b1p + 8);

    // ---- A fragment: fp32 from LDS (contiguous K: lane half h holds K 16h..16h+15), cvt to f16 ----
    const float* fp = &sA[cur][fragRowF + kh * 16];
    float4 q0 = *(const float4*)(fp + 0);
    float4 q1 = *(const float4*)(fp + 4);
    float4 q2 = *(const float4*)(fp + 8);
    float4 q3 = *(const float4*)(fp + 12);

    hvec16 a, b0, b1;
    a[0]  = (_Float16)q0.x; a[1]  = (_Float16)q0.y;
    a[2]  = (_Float16)q0.z; a[3]  = (_Float16)q0.w;
    a[4]  = (_Float16)q1.x; a[5]  = (_Float16)q1.y;
    a[6]  = (_Float16)q1.z; a[7]  = (_Float16)q1.w;
    a[8]  = (_Float16)q2.x; a[9]  = (_Float16)q2.y;
    a[10] = (_Float16)q2.z; a[11] = (_Float16)q2.w;
    a[12] = (_Float16)q3.x; a[13] = (_Float16)q3.y;
    a[14] = (_Float16)q3.z; a[15] = (_Float16)q3.w;
#pragma unroll
    for (int i = 0; i < 8; ++i) {
      b0[i] = b0lo[i]; b0[i + 8] = b0hi[i];
      b1[i] = b1lo[i]; b1[i + 8] = b1hi[i];
    }
    acc0 = __builtin_amdgcn_wmma_f32_16x16x32_f16(false, a, false, b0, (short)0,
                                                  acc0, false, false);
    acc1 = __builtin_amdgcn_wmma_f32_16x16x32_f16(false, a, false, b1, (short)0,
                                                  acc1, false, false);
  }

  // ---- store C (16x16 f32 C/D layout: VGPR r -> M = r + 8*(lane/16)) ----
  const int mrow0 = rowBlk + wm * 16 + kh * 8;
  const int col0  = blockIdx.y * 64 + wn * 32 + (lane & 15);
#pragma unroll
  for (int r = 0; r < 8; ++r) {
    int gr = mrow0 + r;
    if (gr < M) {
#pragma unroll
      for (int s = 0; s < 2; ++s) {
        int gc = col0 + s * 16;
        float v = (s ? acc1[r] : acc0[r]) + bias[gc];
        if (res) v += res[(size_t)gr * N + gc];
        if (Ch) Ch[(size_t)gr * N + gc] = (_Float16)v;
        else    Cf[(size_t)gr * N + gc] = v;
      }
    }
  }
}

// ---------------------------------------------------------------------------
// In-place softmax over 32 (level,point) logits per (query, head).
// ---------------------------------------------------------------------------
__global__ __launch_bounds__(256) void sca_softmax32(float* __restrict__ aw) {
  int w = blockIdx.x * 8 + (threadIdx.x >> 5);
  if (w >= NQ_ * NH_) return;
  int lane = threadIdx.x & 31;
  float* p = aw + (size_t)(w >> 3) * 256 + (w & 7) * 32;
  float v = p[lane];
  float m = v;
#pragma unroll
  for (int s = 16; s; s >>= 1) m = fmaxf(m, __shfl_xor(m, s, 32));
  float e = __expf(v - m);
  float sm = e;
#pragma unroll
  for (int s = 16; s; s >>= 1) sm += __shfl_xor(sm, s, 32);
  p[lane] = e / sm;
}

// ---------------------------------------------------------------------------
// Deformable bilinear sampling. One wave per (cam, q, head); lane = channel d.
// v_proj is f16 (60 MB -> fully L2-resident); gathers are 64B contiguous/wave.
// Skips (cam,q) pairs with bev_mask==0 (60% of the work).
// ---------------------------------------------------------------------------
__global__ __launch_bounds__(256) void sca_deform_sample(
    const _Float16* __restrict__ vproj, const float* __restrict__ off0,
    const float* __restrict__ aw, const float* __restrict__ refp,
    const uint8_t* __restrict__ mask, float* __restrict__ outs) {
  int w = blockIdx.x * 8 + (threadIdx.x >> 5);
  if (w >= NCAMS_ * NQ_ * NH_) return;
  int lane = threadIdx.x & 31;
  int h  = w & 7;
  int cq = w >> 3;             // cam*NQ + q
  int q   = cq % NQ_;
  int cam = cq / NQ_;
  if (!mask[cam * NQ_ + q]) return;

  const _Float16* vbase = vproj + (size_t)cam * LTOT_ * 256 + h * 32 + lane;
  const float* offq = off0 + (size_t)q * 512 + h * 64;   // (l,p,xy)
  const float* awq  = aw + (size_t)q * 256 + h * 32;     // (l,p)
  const float* refq = refp + (size_t)q * 8;              // (NZ=4, 2)

  const int Hs[4] = {92, 46, 23, 12};
  const int Ws[4] = {160, 80, 40, 20};
  const int Ls[4] = {0, 14720, 18400, 19320};

  float acc = 0.0f;
#pragma unroll
  for (int l = 0; l < NL_; ++l) {
    const float Wf = (float)Ws[l], Hf = (float)Hs[l];
#pragma unroll
    for (int p = 0; p < NP_; ++p) {
      float rx = refq[(p & 3) * 2 + 0];
      float ry = refq[(p & 3) * 2 + 1];
      float ox = offq[l * 16 + p * 2 + 0];   // ref divides by [W,H], then
      float oy = offq[l * 16 + p * 2 + 1];   // scales by W,H -> add raw
      float aww = awq[l * 8 + p];
      float x = fmaf(rx, Wf, ox) - 0.5f;
      float y = fmaf(ry, Hf, oy) - 0.5f;
      float x0f = floorf(x), y0f = floorf(y);
      float fx = x - x0f, fy = y - y0f;
      int x0 = (int)x0f, y0 = (int)y0f;
#pragma unroll
      for (int dy = 0; dy < 2; ++dy) {
#pragma unroll
        for (int dx = 0; dx < 2; ++dx) {
          int ix = x0 + dx, iy = y0 + dy;
          if (ix >= 0 && ix < Ws[l] && iy >= 0 && iy < Hs[l]) {
            float cw = (dx ? fx : 1.0f - fx) * (dy ? fy : 1.0f - fy);
            size_t idx = (size_t)(Ls[l] + iy * Ws[l] + ix) * 256;
            acc = fmaf((float)vbase[idx], cw * aww, acc);
          }
        }
      }
    }
  }
  outs[(size_t)w * 32 + lane] = acc;   // == (cam*NQ+q)*256 + h*32 + lane
}

// ---------------------------------------------------------------------------
// slots[q,c] = sum_cam mask0[cam,q]*outs[cam,q,c] / max(1, sum_cam mask)
// ---------------------------------------------------------------------------
__global__ __launch_bounds__(256) void sca_mask_reduce(
    const float* __restrict__ outs, const uint8_t* __restrict__ mask,
    float* __restrict__ slots) {
  int q = blockIdx.x;
  int c = threadIdx.x;
  float s = 0.0f;
  int cnt = 0;
#pragma unroll
  for (int cam = 0; cam < NCAMS_; ++cam) {
    if (mask[cam * NQ_ + q]) {
      cnt++;
      s += outs[(size_t)(cam * NQ_ + q) * 256 + c];
    }
  }
  float cf = cnt > 0 ? (float)cnt : 1.0f;
  slots[(size_t)q * 256 + c] = s / cf;
}

// ---------------------------------------------------------------------------
extern "C" void kernel_launch(void* const* d_in, const int* in_sizes, int n_in,
                              void* d_out, int out_size, void* d_ws, size_t ws_size,
                              hipStream_t stream) {
  (void)in_sizes; (void)n_in; (void)out_size; (void)ws_size;

  const float*   query = (const float*)d_in[0];
  const float*   value = (const float*)d_in[2];
  const float*   qpos  = (const float*)d_in[3];
  const float*   refp  = (const float*)d_in[4];
  const uint8_t* mask  = (const uint8_t*)d_in[5];
  const float*   Wv    = (const float*)d_in[8];
  const float*   bv    = (const float*)d_in[9];
  const float*   Woff  = (const float*)d_in[10];
  const float*   boff  = (const float*)d_in[11];
  const float*   Wat   = (const float*)d_in[12];
  const float*   bat   = (const float*)d_in[13];
  const float*   Wout  = (const float*)d_in[14];
  const float*   bout  = (const float*)d_in[15];
  float* out = (float*)d_out;
  char*  ws  = (char*)d_ws;

  size_t cur = 0;
  auto carve = [&](size_t bytes) -> char* {
    char* r = ws + cur;
    cur = (cur + bytes + 255) & ~(size_t)255;
    return r;
  };
  _Float16* pWv   = (_Float16*)carve((size_t)256 * 256 * 2);
  _Float16* pWoff = (_Float16*)carve((size_t)512 * 256 * 2);
  _Float16* pWat  = (_Float16*)carve((size_t)256 * 256 * 2);
  _Float16* pWout = (_Float16*)carve((size_t)256 * 256 * 2);
  _Float16* vproj = (_Float16*)carve((size_t)NCAMS_ * LTOT_ * 256 * 2);  // 60 MB
  float* qsum  = (float*)carve((size_t)NQ_ * 256 * 4);
  float* off0  = (float*)carve((size_t)NQ_ * 512 * 4);
  float* awp   = (float*)carve((size_t)NQ_ * 256 * 4);
  float* outs  = (float*)carve((size_t)NCAMS_ * NQ_ * 256 * 4);
  float* slots = (float*)carve((size_t)NQ_ * 256 * 4);

  // 1) pack weights to f16 fragment order; fuse query+query_pos
  sca_pack_w<<<16, 256, 0, stream>>>(Wv, pWv, 256);
  sca_pack_w<<<32, 256, 0, stream>>>(Woff, pWoff, 512);
  sca_pack_w<<<16, 256, 0, stream>>>(Wat, pWat, 256);
  sca_pack_w<<<16, 256, 0, stream>>>(Wout, pWout, 256);
  sca_add_qpos<<<(NQ_ * 256 + 255) / 256, 256, 0, stream>>>(query, qpos, qsum,
                                                            NQ_ * 256);

  // 2) value projection: (117360, 256) x (256, 256) -> f16 v_proj
  const int Mv = NCAMS_ * LTOT_;
  sca_gemm_wmma<<<dim3((Mv + 63) / 64, 4), 256, 0, stream>>>(
      value, pWv, bv, nullptr, nullptr, vproj, Mv, 256);

  // 3) query projections (camera-independent: q_all is a broadcast of q)
  sca_gemm_wmma<<<dim3((NQ_ + 63) / 64, 8), 256, 0, stream>>>(
      qsum, pWoff, boff, nullptr, off0, nullptr, NQ_, 512);
  sca_gemm_wmma<<<dim3((NQ_ + 63) / 64, 4), 256, 0, stream>>>(
      qsum, pWat, bat, nullptr, awp, nullptr, NQ_, 256);

  // 4) softmax over 32 per (q,h)
  sca_softmax32<<<(NQ_ * NH_ + 7) / 8, 256, 0, stream>>>(awp);

  // 5) deformable sampling (mask-gated)
  const int nwaves = NCAMS_ * NQ_ * NH_;
  sca_deform_sample<<<(nwaves + 7) / 8, 256, 0, stream>>>(
      vproj, off0, awp, refp, mask, outs);

  // 6) masked mean over cameras
  sca_mask_reduce<<<NQ_, 256, 0, stream>>>(outs, mask, slots);

  // 7) output projection + residual: out = slots @ W_out.T + b_out + query
  sca_gemm_wmma<<<dim3((NQ_ + 63) / 64, 4), 256, 0, stream>>>(
      slots, pWout, bout, query, out, nullptr, NQ_, 256);
}
